// ThoughtMixingHead_61675730370659
// MI455X (gfx1250) — matrix operand order repaired and checked by reading, not compiled, not run on hardware
//
#include <hip/hip_runtime.h>
#include <hip/hip_bf16.h>

// ------------------------------------------------------------------
// Problem constants (B=2, S=2048, D=1024, NH=8, DH=128, DFF=4096, V=32000)
// ------------------------------------------------------------------
#define TOK    4096     // B*S
#define SEQ    2048
#define DM     1024
#define DFF_   4096
#define NHEAD  8
#define DHEAD  128
#define VOCAB  32000

typedef __attribute__((ext_vector_type(16))) __bf16 v16bf;
typedef __attribute__((ext_vector_type(8)))  __bf16 v8bf;
typedef __attribute__((ext_vector_type(4)))  __bf16 v4bf;
typedef __attribute__((ext_vector_type(8)))  float  v8f;
typedef unsigned int u32x4 __attribute__((ext_vector_type(4)));
typedef int          i32x4 __attribute__((ext_vector_type(4)));
typedef int          i32x8 __attribute__((ext_vector_type(8)));

__device__ __forceinline__ v8f wmma_bf16(v16bf a, v16bf b, v8f c) {
  // D = A(16x32 bf16) * B(32x16 bf16) + C(16x16 f32)
  return __builtin_amdgcn_wmma_f32_16x16x32_bf16(false, a, false, b,
                                                 (short)0, c, false, false);
}

// A-fragment (16x32 bf16, M=lane%16): lane half h holds K = [16h..16h+15]
__device__ __forceinline__ v16bf load_afrag(const __bf16* rowk, int hf) {
  v8bf lo = *(const v8bf*)(rowk + hf * 16);
  v8bf hi = *(const v8bf*)(rowk + hf * 16 + 8);
  v16bf r;
#pragma unroll
  for (int i = 0; i < 8; ++i) { r[i] = lo[i]; r[i + 8] = hi[i]; }
  return r;
}

// ------------------------------------------------------------------
// Tensor Data Mover: 2D tile (global -> LDS), rows of row_len_dw DWORDs,
// row stride stride_dw DWORDs, LDS rows padded by 4 DWORDs (8 bf16).
// D# per CDNA5 ISA ch.8: group0 {count=1, lds_addr, global_addr, type=2},
// group1 {data_size=4B, pad_enable, pad_interval, pad_amount=4dw, dims/strides}.
// pad_interval_code: 3 -> every 16 DWORDs, 5 -> every 64 DWORDs.
// ------------------------------------------------------------------
__device__ __forceinline__ void tdm_load_2d(unsigned ldsOff, const void* gptr,
                                            int tile_rows, int row_len_dw,
                                            int stride_dw, int pad_interval_code) {
#if __has_builtin(__builtin_amdgcn_tensor_load_to_lds)
  unsigned long long ga = (unsigned long long)gptr;
  u32x4 g0;
  g0[0] = 1u;                                            // count=1 (valid)
  g0[1] = ldsOff;                                        // LDS byte address
  g0[2] = (unsigned)(ga & 0xffffffffu);                  // global_addr[31:0]
  g0[3] = (unsigned)((ga >> 32) & 0x1ffffffu) | (2u << 30); // addr[56:32] | type=2
  i32x8 g1;
  g1[0] = (int)((2u << 16) | (1u << 20) |
                ((unsigned)pad_interval_code << 22) | (3u << 25));
  g1[1] = (int)((unsigned)row_len_dw << 16);             // tensor_dim0[15:0]
  g1[2] = (int)((unsigned)tile_rows << 16);              // tensor_dim1[15:0]
  g1[3] = (int)((unsigned)row_len_dw << 16);             // tile_dim0
  g1[4] = tile_rows;                                     // tile_dim1 (tile_dim2=0)
  g1[5] = stride_dw;                                     // tensor_dim0_stride[31:0]
  g1[6] = 0;
  g1[7] = 0;
  i32x4 z4 = {0, 0, 0, 0};
#if defined(__clang_major__) && __clang_major__ >= 23
  i32x8 z8 = {0, 0, 0, 0, 0, 0, 0, 0};
  __builtin_amdgcn_tensor_load_to_lds(g0, g1, z4, z4, z8, 0);
#else
  __builtin_amdgcn_tensor_load_to_lds(g0, g1, z4, z4, 0);
#endif
#endif
}

__device__ __forceinline__ void tdm_wait() {
#if __has_builtin(__builtin_amdgcn_s_wait_tensorcnt)
  __builtin_amdgcn_s_wait_tensorcnt(0);
#endif
}

__device__ __forceinline__ unsigned lds_off(const void* p) {
  return (unsigned)(unsigned long long)p;   // addr[31:0] == LDS offset
}

// ------------------------------------------------------------------
// thought mask: open-span scan + closed-span (trailing eot) check
// ------------------------------------------------------------------
__global__ void tm_mask_kernel(const int* __restrict__ ids,
                               const int* __restrict__ sotp,
                               const int* __restrict__ eotp,
                               float* __restrict__ tmaskf) {
  int b = threadIdx.x;
  if (b >= 2) return;
  const int sot = *sotp, eot = *eotp;
  const int* r = ids + b * SEQ;
  float* o = tmaskf + b * SEQ;
  bool s = false;
  for (int t = 0; t < SEQ; ++t) {
    bool iso = (r[t] == sot), ieo = (r[t] == eot);
    bool open = iso || s;
    s = iso || (s && !ieo);
    o[t] = open ? 1.0f : 0.0f;
  }
  bool anyE = false;
  for (int t = SEQ - 1; t >= 0; --t) {
    anyE = anyE || (r[t] == eot);
    if (!anyE) o[t] = 0.0f;
  }
}

// ------------------------------------------------------------------
// fp32 -> bf16 conversion (vectorized x4)
// ------------------------------------------------------------------
__global__ __launch_bounds__(256) void tm_cvt_kernel(const float* __restrict__ in,
                                                     __bf16* __restrict__ out, int n4) {
  int i = blockIdx.x * 256 + threadIdx.x;
  if (i >= n4) return;
  float4 f = ((const float4*)in)[i];
  v4bf o;
  o[0] = (__bf16)f.x; o[1] = (__bf16)f.y; o[2] = (__bf16)f.z; o[3] = (__bf16)f.w;
  ((v4bf*)out)[i] = o;
}

// copy xb (bf16 [TOK,DM]) into left half of concat buffer cb (bf16 [TOK,2*DM])
__global__ __launch_bounds__(256) void tm_cbleft_kernel(const __bf16* __restrict__ xb,
                                                        __bf16* __restrict__ cb) {
  int i = blockIdx.x * 256 + threadIdx.x;            // over TOK*DM/4
  if (i >= TOK * (DM / 4)) return;
  int m = i >> 8;                                    // DM/4 == 256 per row
  int j4 = i & 255;
  ((v4bf*)cb)[(size_t)m * (2 * DM / 4) + j4] = ((const v4bf*)xb)[i];
}

// ------------------------------------------------------------------
// Generic bf16 WMMA GEMM:  C[M,N] = A[M,K] * W[N,K]^T  (+epilogue)
// block = 256 threads = 8 waves (2 M x 4 N), wave tile 32x32, block tile 64x128.
// Block tiles are TDM-staged into LDS (rows padded +8 bf16 -> conflict-free
// b128 reads) and double-buffered; wave 0 drives the DMA, one barrier/step.
// ------------------------------------------------------------------
#define LPAD 40   // 32 elems + 8 pad

struct GemmParams {
  const __bf16* A;
  const __bf16* W;
  int N, K;
  const float* bias;    // [N]
  const float* resid;   // [M,N] fp32 (epi 1)
  float*   outF;        // fp32 out (epi 1,3)
  __bf16*  outB;        // bf16 out (epi 2)
  __bf16 *qb, *kb, *vtb; // epi 0 (QKV split)
};

template <int EPI>
__device__ __forceinline__ void epi_store(const GemmParams& p, float c, int m, int n) {
  if constexpr (EPI == 0) {                 // QKV: +bias, split q/k, v transposed
    float v = c + p.bias[n];
    if (n < DM) {
      p.qb[(size_t)m * DM + n] = (__bf16)v;
    } else if (n < 2 * DM) {
      p.kb[(size_t)m * DM + (n - DM)] = (__bf16)v;
    } else {
      int d = n - 2 * DM;
      int b = m >> 11, s = m & (SEQ - 1);
      // vt layout: [B, NH, DHEAD, SEQ]
      p.vtb[((size_t)(b * NHEAD + (d >> 7)) * DHEAD + (d & (DHEAD - 1))) * SEQ + s] =
          (__bf16)v;
    }
  } else if constexpr (EPI == 1) {          // + bias + fp32 residual -> fp32
    p.outF[(size_t)m * p.N + n] = c + p.bias[n] + p.resid[(size_t)m * p.N + n];
  } else if constexpr (EPI == 2) {          // relu(+bias) -> bf16
    p.outB[(size_t)m * p.N + n] = (__bf16)fmaxf(c + p.bias[n], 0.0f);
  } else {                                  // + bias -> fp32
    p.outF[(size_t)m * p.N + n] = c + p.bias[n];
  }
}

template <int EPI>
__global__ __launch_bounds__(256) void tm_gemm_kernel(GemmParams p) {
  __shared__ __bf16 Alds[2][64 * LPAD];
  __shared__ __bf16 Blds[2][128 * LPAD];
  const int lane = threadIdx.x & 31;
  const int wave = threadIdx.x >> 5;
  const int lr = lane & 15, hf = lane >> 4;
  const int wm = wave & 1, wn = wave >> 1;
  const int m0 = blockIdx.y * 64;
  const int n0 = blockIdx.x * 128;
  const int K = p.K;
  const int nsteps = K >> 5;

  auto issue = [&](int buf, int k) {
    tdm_load_2d(lds_off(&Alds[buf][0]), p.A + (size_t)m0 * K + k, 64, 16, K >> 1, 3);
    tdm_load_2d(lds_off(&Blds[buf][0]), p.W + (size_t)n0 * K + k, 128, 16, K >> 1, 3);
  };

  if (wave == 0) issue(0, 0);

  v8f c00 = {}, c01 = {}, c10 = {}, c11 = {};
  for (int s = 0; s < nsteps; ++s) {
    const int buf = s & 1;
    if (wave == 0) tdm_wait();          // buf's DMA complete
    __syncthreads();                    // publish; prev compute drained
    if (wave == 0 && s + 1 < nsteps) issue(buf ^ 1, (s + 1) << 5);
    const __bf16* Ab = &Alds[buf][0];
    const __bf16* Bb = &Blds[buf][0];
    v16bf a0 = load_afrag(Ab + (wm * 32 + lr) * LPAD, hf);
    v16bf a1 = load_afrag(Ab + (wm * 32 + 16 + lr) * LPAD, hf);
    v16bf b0 = *(const v16bf*)(Bb + (wn * 32 + lr) * LPAD + hf * 16);
    v16bf b1 = *(const v16bf*)(Bb + (wn * 32 + 16 + lr) * LPAD + hf * 16);
    c00 = wmma_bf16(a0, b0, c00);
    c01 = wmma_bf16(a0, b1, c01);
    c10 = wmma_bf16(a1, b0, c10);
    c11 = wmma_bf16(a1, b1, c11);
  }

  const int rb = m0 + wm * 32 + hf * 8;
  const int nb = n0 + wn * 32;
#pragma unroll
  for (int r = 0; r < 8; ++r) {
    epi_store<EPI>(p, c00[r], rb + r,      nb + lr);
    epi_store<EPI>(p, c01[r], rb + r,      nb + 16 + lr);
    epi_store<EPI>(p, c10[r], rb + 16 + r, nb + lr);
    epi_store<EPI>(p, c11[r], rb + 16 + r, nb + 16 + lr);
  }
}

// ------------------------------------------------------------------
// Flash attention with key masking (thought keys only; -1e9 like reference)
// grid = (SEQ/64, B*NH), block = 128 (4 waves), each wave = 16 query rows.
// The 32-key K tile (32x128) and V^T tile (128x32) are shared by all 4 waves:
// TDM-staged into LDS, double-buffered, one barrier per key tile.
// ------------------------------------------------------------------
__global__ __launch_bounds__(128) void tm_flash_kernel(const __bf16* __restrict__ qb,
                                                       const __bf16* __restrict__ kb,
                                                       const __bf16* __restrict__ vtb,
                                                       const float* __restrict__ tmaskf,
                                                       __bf16* __restrict__ ctxb) {
  __shared__ __bf16 Klds[2][32][DHEAD + 8];   // +8 bf16 row pad
  __shared__ __bf16 Vlds[2][DHEAD][40];       // 32 + 8 pad
  __shared__ __bf16 plds[4][16 * 32];
  const int lane = threadIdx.x & 31, wave = threadIdx.x >> 5;
  const int lr = lane & 15, hf = lane >> 4;
  const int q0 = blockIdx.x * 64 + wave * 16;
  const int bh = blockIdx.y;
  const int b = bh >> 3, h = bh & 7;
  const float scale = 0.08838834764831845f; // 1/sqrt(128)

  const __bf16* kbase0 = kb + (size_t)(b * SEQ) * DM + h * DHEAD;
  const __bf16* vbase0 = vtb + (size_t)(b * NHEAD + h) * DHEAD * SEQ;
  auto issueKV = [&](int buf, int kt) {
    tdm_load_2d(lds_off(&Klds[buf][0][0]), kbase0 + (size_t)kt * DM, 32, 64, DM >> 1, 5);
    tdm_load_2d(lds_off(&Vlds[buf][0][0]), vbase0 + kt, DHEAD, 16, SEQ >> 1, 3);
  };
  if (wave == 0) issueKV(0, 0);

  // Q fragments: 4 d-tiles of 32 (held in registers for the whole pass)
  const __bf16* qrow = qb + ((size_t)(b * SEQ + q0 + lr) * DM + h * DHEAD);
  v16bf aq[4];
#pragma unroll
  for (int j = 0; j < 4; ++j) aq[j] = load_afrag(qrow + j * 32, hf);

  v8f o[8] = {};
  float mrow[8], lrow[8];
#pragma unroll
  for (int r = 0; r < 8; ++r) { mrow[r] = -1e30f; lrow[r] = 0.0f; }

  const float* mbase = tmaskf + b * SEQ;
  __bf16* myl = plds[wave];

  for (int kt = 0; kt < SEQ; kt += 32) {
    const int buf = (kt >> 5) & 1;
    if (wave == 0) tdm_wait();
    __syncthreads();
    if (wave == 0 && kt + 32 < SEQ) issueKV(buf ^ 1, kt + 32);

    // ---- S = Q * K^T for 32 keys (two 16-key C tiles), K tile from LDS
    v8f sc0 = {}, sc1 = {};
#pragma unroll
    for (int j = 0; j < 4; ++j) {
      v16bf bk0 = *(const v16bf*)(&Klds[buf][lr][j * 32 + hf * 16]);
      v16bf bk1 = *(const v16bf*)(&Klds[buf][16 + lr][j * 32 + hf * 16]);
      sc0 = wmma_bf16(aq[j], bk0, sc0);
      sc1 = wmma_bf16(aq[j], bk1, sc1);
    }
    const float mk0 = (mbase[kt + lr] - 1.0f) * 1e9f;        // 0 or -1e9
    const float mk1 = (mbase[kt + 16 + lr] - 1.0f) * 1e9f;

    // ---- online softmax per row (row r + 8*hf; 16 keys per half-wave)
#pragma unroll
    for (int r = 0; r < 8; ++r) {
      float s0 = sc0[r] * scale + mk0;
      float s1 = sc1[r] * scale + mk1;
      float mx = fmaxf(s0, s1);
#pragma unroll
      for (int off = 1; off < 16; off <<= 1) mx = fmaxf(mx, __shfl_xor(mx, off, 32));
      float nm = fmaxf(mrow[r], mx);
      float alpha = __expf(mrow[r] - nm);
      mrow[r] = nm;
      float p0 = __expf(s0 - nm);
      float p1 = __expf(s1 - nm);
      float ps = p0 + p1;
#pragma unroll
      for (int off = 1; off < 16; off <<= 1) ps += __shfl_xor(ps, off, 32);
      lrow[r] = lrow[r] * alpha + ps;
#pragma unroll
      for (int jd = 0; jd < 8; ++jd) o[jd][r] = o[jd][r] * alpha;
      const int row = r + hf * 8;
      myl[row * 32 + lr]      = (__bf16)p0;
      myl[row * 32 + 16 + lr] = (__bf16)p1;
    }

    // ---- O += P(16x32) * V(32x128); P re-laid out through LDS as A-fragment
    v16bf ap = load_afrag(myl + lr * 32, hf);
#pragma unroll
    for (int jd = 0; jd < 8; ++jd) {
      v16bf bv = *(const v16bf*)(&Vlds[buf][jd * 16 + lr][hf * 16]);
      o[jd] = wmma_bf16(ap, bv, o[jd]);
    }
  }

  // ---- epilogue: divide by row sums, write ctx bf16 [TOK, DM]
#pragma unroll
  for (int jd = 0; jd < 8; ++jd) {
#pragma unroll
    for (int r = 0; r < 8; ++r) {
      float v = o[jd][r] / lrow[r];
      int row = q0 + r + hf * 8;
      ctxb[(size_t)(b * SEQ + row) * DM + h * DHEAD + jd * 16 + lr] = (__bf16)v;
    }
  }
}

// ------------------------------------------------------------------
// LayerNorm kernels (one row of DM=1024 per block, 256 threads x 4 elems)
// ------------------------------------------------------------------
__global__ __launch_bounds__(256) void tm_ln1_kernel(const float* __restrict__ in,
                                                     const float* __restrict__ g,
                                                     const float* __restrict__ bb,
                                                     float* __restrict__ outF,
                                                     __bf16* __restrict__ outB) {
  __shared__ float sA[256], sB[256];
  const int row = blockIdx.x, tid = threadIdx.x;
  const float* x = in + (size_t)row * DM;
  float v[4], s = 0.0f, s2 = 0.0f;
#pragma unroll
  for (int i = 0; i < 4; ++i) {
    v[i] = x[tid + i * 256];
    s += v[i]; s2 += v[i] * v[i];
  }
  sA[tid] = s; sB[tid] = s2;
  __syncthreads();
  for (int off = 128; off > 0; off >>= 1) {
    if (tid < off) { sA[tid] += sA[tid + off]; sB[tid] += sB[tid + off]; }
    __syncthreads();
  }
  const float mu = sA[0] * (1.0f / DM);
  const float var = sB[0] * (1.0f / DM) - mu * mu;
  const float rstd = rsqrtf(var + 1e-5f);
#pragma unroll
  for (int i = 0; i < 4; ++i) {
    int j = tid + i * 256;
    float y = (v[i] - mu) * rstd * g[j] + bb[j];
    outF[(size_t)row * DM + j] = y;
    outB[(size_t)row * DM + j] = (__bf16)y;
  }
}

// LN2 + thought scatter: te = tmask ? LN(r2) : x ; write into right half of cb
__global__ __launch_bounds__(256) void tm_ln2_kernel(const float* __restrict__ in,
                                                     const float* __restrict__ g,
                                                     const float* __restrict__ bb,
                                                     const float* __restrict__ tmaskf,
                                                     const float* __restrict__ hs,
                                                     __bf16* __restrict__ cb) {
  __shared__ float sA[256], sB[256];
  const int row = blockIdx.x, tid = threadIdx.x;
  const float* x = in + (size_t)row * DM;
  float v[4], s = 0.0f, s2 = 0.0f;
#pragma unroll
  for (int i = 0; i < 4; ++i) {
    v[i] = x[tid + i * 256];
    s += v[i]; s2 += v[i] * v[i];
  }
  sA[tid] = s; sB[tid] = s2;
  __syncthreads();
  for (int off = 128; off > 0; off >>= 1) {
    if (tid < off) { sA[tid] += sA[tid + off]; sB[tid] += sB[tid + off]; }
    __syncthreads();
  }
  const float mu = sA[0] * (1.0f / DM);
  const float var = sB[0] * (1.0f / DM) - mu * mu;
  const float rstd = rsqrtf(var + 1e-5f);
  const bool t = tmaskf[row] > 0.5f;
#pragma unroll
  for (int i = 0; i < 4; ++i) {
    int j = tid + i * 256;
    float y = (v[i] - mu) * rstd * g[j] + bb[j];
    float te = t ? y : hs[(size_t)row * DM + j];
    cb[(size_t)row * (2 * DM) + DM + j] = (__bf16)te;
  }
}

// ------------------------------------------------------------------
// gate + mix: one wave per token
// ------------------------------------------------------------------
__global__ __launch_bounds__(128) void tm_mix_kernel(const __bf16* __restrict__ cb,
                                                     const float* __restrict__ Wg,
                                                     const float* __restrict__ bg,
                                                     const float* __restrict__ tmaskf,
                                                     const float* __restrict__ hs,
                                                     const float* __restrict__ mcf,
                                                     __bf16* __restrict__ mixedb) {
  const int lane = threadIdx.x & 31, wave = threadIdx.x >> 5;
  const int m = blockIdx.x * 4 + wave;
  const __bf16* te = cb + (size_t)m * (2 * DM) + DM;
  float gacc = 0.0f;
  for (int j = lane; j < DM; j += 32) gacc += (float)te[j] * Wg[j];
#pragma unroll
  for (int off = 1; off < 32; off <<= 1) gacc += __shfl_xor(gacc, off, 32);
  float gate = 1.0f / (1.0f + __expf(-(gacc + bg[0])));
  float eff = (tmaskf[m] > 0.5f) ? gate : 0.1f;
  for (int j = lane; j < DM; j += 32) {
    float mv = (1.0f - eff) * hs[(size_t)m * DM + j] + eff * mcf[(size_t)m * DM + j];
    mixedb[(size_t)m * DM + j] = (__bf16)mv;
  }
}

// ------------------------------------------------------------------
// host: orchestrate the pipeline (workspace ~258.5 MiB)
// ------------------------------------------------------------------
extern "C" void kernel_launch(void* const* d_in, const int* in_sizes, int n_in,
                              void* d_out, int out_size, void* d_ws, size_t ws_size,
                              hipStream_t stream) {
  const float* hs    = (const float*)d_in[0];
  const int*   ids   = (const int*)d_in[1];
  const int*   sotp  = (const int*)d_in[2];
  const int*   eotp  = (const int*)d_in[3];
  const float* Wqkv  = (const float*)d_in[4];
  const float* bqkv  = (const float*)d_in[5];
  const float* Wo    = (const float*)d_in[6];
  const float* bo    = (const float*)d_in[7];
  const float* ln1g  = (const float*)d_in[8];
  const float* ln1b  = (const float*)d_in[9];
  const float* W1    = (const float*)d_in[10];
  const float* b1    = (const float*)d_in[11];
  const float* W2    = (const float*)d_in[12];
  const float* b2    = (const float*)d_in[13];
  const float* ln2g  = (const float*)d_in[14];
  const float* ln2b  = (const float*)d_in[15];
  const float* Wg    = (const float*)d_in[16];
  const float* bg    = (const float*)d_in[17];
  const float* Wm    = (const float*)d_in[18];
  const float* bm    = (const float*)d_in[19];
  const float* Wout  = (const float*)d_in[20];
  const float* bout  = (const float*)d_in[21];
  float* out = (float*)d_out;

  // ---- workspace carving
  char* w = (char*)d_ws;
  auto alloc = [&](size_t bytes) -> void* {
    void* p = (void*)w;
    w += (bytes + 255) & ~(size_t)255;
    return p;
  };
  float*  tmaskf = (float*) alloc((size_t)TOK * 4);
  __bf16* xb     = (__bf16*)alloc((size_t)TOK * DM * 2);
  __bf16* wqkvb  = (__bf16*)alloc((size_t)3 * DM * DM * 2);
  __bf16* wob    = (__bf16*)alloc((size_t)DM * DM * 2);
  __bf16* w1b    = (__bf16*)alloc((size_t)DFF_ * DM * 2);
  __bf16* w2b    = (__bf16*)alloc((size_t)DM * DFF_ * 2);
  __bf16* wmb    = (__bf16*)alloc((size_t)DM * 2 * DM * 2);
  __bf16* woutb  = (__bf16*)alloc((size_t)VOCAB * DM * 2);
  __bf16* qb     = (__bf16*)alloc((size_t)TOK * DM * 2);
  __bf16* kb     = (__bf16*)alloc((size_t)TOK * DM * 2);
  __bf16* vtb    = (__bf16*)alloc((size_t)TOK * DM * 2);
  __bf16* ctxb   = (__bf16*)alloc((size_t)TOK * DM * 2);
  float*  r1f    = (float*) alloc((size_t)TOK * DM * 4);
  float*  x1f    = (float*) alloc((size_t)TOK * DM * 4);
  __bf16* x1b    = (__bf16*)alloc((size_t)TOK * DM * 2);
  __bf16* hb     = (__bf16*)alloc((size_t)TOK * DFF_ * 2);
  float*  r2f    = (float*) alloc((size_t)TOK * DM * 4);
  __bf16* cb     = (__bf16*)alloc((size_t)TOK * 2 * DM * 2);
  float*  mcf    = (float*) alloc((size_t)TOK * DM * 4);
  __bf16* mixedb = (__bf16*)alloc((size_t)TOK * DM * 2);

  // ---- 1) thought mask
  tm_mask_kernel<<<1, 32, 0, stream>>>(ids, sotp, eotp, tmaskf);

  // ---- 2) fp32 -> bf16 conversions
  auto cvt = [&](const float* src, __bf16* dst, size_t n) {
    int n4 = (int)(n / 4);
    tm_cvt_kernel<<<(n4 + 255) / 256, 256, 0, stream>>>(src, dst, n4);
  };
  cvt(hs,   xb,    (size_t)TOK * DM);
  cvt(Wqkv, wqkvb, (size_t)3 * DM * DM);
  cvt(Wo,   wob,   (size_t)DM * DM);
  cvt(W1,   w1b,   (size_t)DFF_ * DM);
  cvt(W2,   w2b,   (size_t)DM * DFF_);
  cvt(Wm,   wmb,   (size_t)DM * 2 * DM);
  cvt(Wout, woutb, (size_t)VOCAB * DM);
  tm_cbleft_kernel<<<(TOK * (DM / 4) + 255) / 256, 256, 0, stream>>>(xb, cb);

  // ---- 3) QKV projection (epilogue: bias + split + V transpose)
  {
    GemmParams p{};
    p.A = xb; p.W = wqkvb; p.N = 3 * DM; p.K = DM; p.bias = bqkv;
    p.qb = qb; p.kb = kb; p.vtb = vtb;
    tm_gemm_kernel<0><<<dim3((3 * DM) / 128, TOK / 64), 256, 0, stream>>>(p);
  }

  // ---- 4) masked flash attention
  tm_flash_kernel<<<dim3(SEQ / 64, 2 * NHEAD), 128, 0, stream>>>(qb, kb, vtb, tmaskf, ctxb);

  // ---- 5) Wo projection + residual -> r1 (fp32)
  {
    GemmParams p{};
    p.A = ctxb; p.W = wob; p.N = DM; p.K = DM; p.bias = bo; p.resid = hs; p.outF = r1f;
    tm_gemm_kernel<1><<<dim3(DM / 128, TOK / 64), 256, 0, stream>>>(p);
  }

  // ---- 6) LN1 -> x1 (fp32 + bf16)
  tm_ln1_kernel<<<TOK, 256, 0, stream>>>(r1f, ln1g, ln1b, x1f, x1b);

  // ---- 7) FFN up + relu -> hb (bf16)
  {
    GemmParams p{};
    p.A = x1b; p.W = w1b; p.N = DFF_; p.K = DM; p.bias = b1; p.outB = hb;
    tm_gemm_kernel<2><<<dim3(DFF_ / 128, TOK / 64), 256, 0, stream>>>(p);
  }

  // ---- 8) FFN down + residual -> r2 (fp32)
  {
    GemmParams p{};
    p.A = hb; p.W = w2b; p.N = DM; p.K = DFF_; p.bias = b2; p.resid = x1f; p.outF = r2f;
    tm_gemm_kernel<1><<<dim3(DM / 128, TOK / 64), 256, 0, stream>>>(p);
  }

  // ---- 9) LN2 + thought scatter -> right half of cb (bf16)
  tm_ln2_kernel<<<TOK, 256, 0, stream>>>(r2f, ln2g, ln2b, tmaskf, hs, cb);

  // ---- 10) mixer projection: mc = [x, te] @ Wm^T + bm (fp32)
  {
    GemmParams p{};
    p.A = cb; p.W = wmb; p.N = DM; p.K = 2 * DM; p.bias = bm; p.outF = mcf;
    tm_gemm_kernel<3><<<dim3(DM / 128, TOK / 64), 256, 0, stream>>>(p);
  }

  // ---- 11) gate + mix -> mixedb (bf16)
  tm_mix_kernel<<<TOK / 4, 128, 0, stream>>>(cb, Wg, bg, tmaskf, hs, mcf, mixedb);

  // ---- 12) vocab head: logits = mixed @ Wout^T + bout (fp32 -> d_out)
  {
    GemmParams p{};
    p.A = mixedb; p.W = woutb; p.N = VOCAB; p.K = DM; p.bias = bout; p.outF = out;
    tm_gemm_kernel<3><<<dim3(VOCAB / 128, TOK / 64), 256, 0, stream>>>(p);
  }
}
